// CrossAttention_33045478375521
// MI455X (gfx1250) — matrix operand order repaired and checked by reading, not compiled, not run on hardware
//
#include <hip/hip_runtime.h>
#include <hip/hip_bf16.h>

// CDNA5 / gfx1250 cross-attention, bf16 WMMA (v_wmma_f32_16x16x32_bf16),
// flash-style online softmax (attention matrix never touches HBM).

typedef __attribute__((ext_vector_type(16))) __bf16 v16bf;
typedef __attribute__((ext_vector_type(8)))  __bf16 v8bf;
typedef __attribute__((ext_vector_type(8)))  float  v8f;

#define BATCH_ 4
#define SEQ_   2048   // N and NTOK
#define CDIM_  1024
#define HEADS_ 16
#define HD_    64

__device__ __forceinline__ __bf16 f2bf(float f) {
  unsigned u = __builtin_bit_cast(unsigned, f);
  u += 0x7FFFu + ((u >> 16) & 1u);   // round-to-nearest-even
  unsigned short h = (unsigned short)(u >> 16);
  return __builtin_bit_cast(__bf16, h);
}

__device__ __forceinline__ v8f vzero8() {
  union { v8f v; float e[8]; } u;
#pragma unroll
  for (int i = 0; i < 8; ++i) u.e[i] = 0.0f;
  return u.v;
}

__device__ __forceinline__ v8f wmma_bf16(v16bf a, v16bf b, v8f c) {
  // (neg_a, A, neg_b, B, c_mod, C, reuse_a, reuse_b)
  return __builtin_amdgcn_wmma_f32_16x16x32_bf16(false, a, false, b, (short)0, c,
                                                 false, false);
}

__device__ __forceinline__ v16bf cat8(v8bf lo, v8bf hi) {
  return __builtin_shufflevector(lo, hi, 0, 1, 2, 3, 4, 5, 6, 7,
                                 8, 9, 10, 11, 12, 13, 14, 15);
}

// ---------------------------------------------------------------- convert
__global__ void cvt_f32_to_bf16_kernel(const float* __restrict__ in,
                                       __bf16* __restrict__ out, unsigned n) {
  unsigned i = blockIdx.x * 256u + threadIdx.x;
  if (i < n) out[i] = f2bf(in[i]);
}

// ------------------------------------------------------------- V transpose
// V: [B][T][H*64] -> Vt: [B][H][64][T]
__global__ void transpose_v_kernel(const __bf16* __restrict__ V,
                                   __bf16* __restrict__ Vt) {
  unsigned i = blockIdx.x * 256u + threadIdx.x;  // [0, B*H*64*T)
  unsigned t = i & 2047u;
  unsigned d = (i >> 11) & 63u;
  unsigned h = (i >> 17) & 15u;
  unsigned b = i >> 21;
  Vt[i] = V[((size_t)(b * SEQ_ + t)) * CDIM_ + h * HD_ + d];
}

// ----------------------------------------------------------------- GEMM
// C[M][Nn] = A[M][K](bf16) @ W[Nn][K](bf16)^T + bias
// block tile 128x128 (8 waves), wave tile 32x64 (2x4 WMMA accumulators)
template <bool OUT_F32>
__device__ __forceinline__ void gemm_body(const __bf16* __restrict__ A,
                                          const __bf16* __restrict__ W,
                                          const float* __restrict__ bias,
                                          void* __restrict__ Cout,
                                          int Nn, int K) {
  const int lane = threadIdx.x & 31;
  const int w    = threadIdx.x >> 5;
  const int half = lane >> 4;
  const int l16  = lane & 15;
  const int mbase = blockIdx.x * 128 + (w >> 1) * 32;
  const int nbase = blockIdx.y * 128 + (w & 1) * 64;

  v8f acc[2][4];
#pragma unroll
  for (int mt = 0; mt < 2; ++mt)
#pragma unroll
    for (int nt = 0; nt < 4; ++nt) acc[mt][nt] = vzero8();

#pragma unroll 2
  for (int k0 = 0; k0 < K; k0 += 32) {
    v16bf a[2];
#pragma unroll
    for (int mt = 0; mt < 2; ++mt) {
      const __bf16* pa =
          A + (size_t)(mbase + mt * 16 + l16) * (size_t)K + k0 + half * 16;
      a[mt] = cat8(*(const v8bf*)pa, *(const v8bf*)(pa + 8));
    }
    v16bf b[4];
#pragma unroll
    for (int nt = 0; nt < 4; ++nt) {
      const __bf16* pb =
          W + (size_t)(nbase + nt * 16 + l16) * (size_t)K + k0 + half * 16;
      b[nt] = *(const v16bf*)pb;
    }
#pragma unroll
    for (int mt = 0; mt < 2; ++mt)
#pragma unroll
      for (int nt = 0; nt < 4; ++nt)
        acc[mt][nt] = wmma_bf16(a[mt], b[nt], acc[mt][nt]);
  }

#pragma unroll
  for (int mt = 0; mt < 2; ++mt)
#pragma unroll
    for (int nt = 0; nt < 4; ++nt) {
      const int col = nbase + nt * 16 + l16;
      const float bv = bias[col];
      union { v8f v; float e[8]; } u;
      u.v = acc[mt][nt];
#pragma unroll
      for (int r = 0; r < 8; ++r) {
        const int row = mbase + mt * 16 + r + half * 8;
        const float val = u.e[r] + bv;
        if (OUT_F32)
          ((float*)Cout)[(size_t)row * Nn + col] = val;
        else
          ((__bf16*)Cout)[(size_t)row * Nn + col] = f2bf(val);
      }
    }
}

__global__ void __launch_bounds__(256)
gemm_bf16out_kernel(const __bf16* __restrict__ A, const __bf16* __restrict__ W,
                    const float* __restrict__ bias, __bf16* __restrict__ C,
                    int Nn, int K) {
  gemm_body<false>(A, W, bias, (void*)C, Nn, K);
}

__global__ void __launch_bounds__(256)
gemm_f32out_kernel(const __bf16* __restrict__ A, const __bf16* __restrict__ W,
                   const float* __restrict__ bias, float* __restrict__ C,
                   int Nn, int K) {
  gemm_body<true>(A, W, bias, (void*)C, Nn, K);
}

// ------------------------------------------------------------- attention
// One wave per (b, h, 16-row m-tile). Flash-style single pass over T in
// chunks of 32 columns: S = Q*K^T (4 WMMA), online softmax (shfl_xor
// butterflies within 16-lane halves), P staged through per-wave LDS to
// convert C-layout -> A-layout, then O += P*V (4 WMMA, V pre-transposed).
__global__ void __launch_bounds__(256)
attn_kernel(const __bf16* __restrict__ Q, const __bf16* __restrict__ Kb,
            const __bf16* __restrict__ Vt, __bf16* __restrict__ Y) {
  __shared__ __bf16 Plds[8][16][32];  // 8 KB, one 16x32 tile per wave

  const int lane = threadIdx.x & 31;
  const int w    = threadIdx.x >> 5;
  const int half = lane >> 4;
  const int l16  = lane & 15;

  const int id = blockIdx.x * 8 + w;   // 0 .. B*H*(SEQ/16)-1
  const int mt = id & 127;
  const int h  = (id >> 7) & 15;
  const int b  = id >> 11;
  const int m0 = mt * 16;

  // Q fragments (16 rows x 64 d), two K-steps of 32
  v16bf qa[2];
#pragma unroll
  for (int ks = 0; ks < 2; ++ks) {
    const __bf16* p = Q + (size_t)(b * SEQ_ + m0 + l16) * CDIM_ + h * HD_ +
                      ks * 32 + half * 16;
    qa[ks] = cat8(*(const v8bf*)p, *(const v8bf*)(p + 8));
  }

  v8f oacc[4];
#pragma unroll
  for (int dt = 0; dt < 4; ++dt) oacc[dt] = vzero8();
  float mrow[8], lrow[8];
#pragma unroll
  for (int r = 0; r < 8; ++r) { mrow[r] = -3.0e38f; lrow[r] = 0.0f; }

  const float scale = 0.125f;  // 1/sqrt(64)

  for (int t0 = 0; t0 < SEQ_; t0 += 32) {
    if (t0 + 32 < SEQ_)  // prefetch next K tile rows (global_prefetch_b8)
      __builtin_prefetch(
          Kb + (size_t)(b * SEQ_ + t0 + 32 + l16) * CDIM_ + h * HD_, 0, 1);

    // ---- S = Q * K^T : 16 x 32 chunk = two 16x16 tiles, K-dim = 64
    v8f sa[2];
    sa[0] = vzero8(); sa[1] = vzero8();
#pragma unroll
    for (int tt = 0; tt < 2; ++tt)
#pragma unroll
      for (int ks = 0; ks < 2; ++ks) {
        const __bf16* p = Kb +
            (size_t)(b * SEQ_ + t0 + tt * 16 + l16) * CDIM_ + h * HD_ +
            ks * 32 + half * 16;
        v16bf kf = *(const v16bf*)p;
        sa[tt] = wmma_bf16(qa[ks], kf, sa[tt]);
      }

    // ---- online softmax (row = acc element r + 8*half, col = lane&15)
    union { v8f v; float e[8]; } s0, s1;
    s0.v = sa[0]; s1.v = sa[1];
    float p0[8], p1[8], corr[8];
#pragma unroll
    for (int r = 0; r < 8; ++r) {
      float a0 = s0.e[r] * scale;
      float a1 = s1.e[r] * scale;
      float mx = fmaxf(a0, a1);
#pragma unroll
      for (int msk = 1; msk < 16; msk <<= 1)
        mx = fmaxf(mx, __shfl_xor(mx, msk, 32));
      const float mnew = fmaxf(mrow[r], mx);
      const float c = __expf(mrow[r] - mnew);
      const float e0 = __expf(a0 - mnew);
      const float e1 = __expf(a1 - mnew);
      float sum = e0 + e1;
#pragma unroll
      for (int msk = 1; msk < 16; msk <<= 1)
        sum += __shfl_xor(sum, msk, 32);
      lrow[r] = lrow[r] * c + sum;
      mrow[r] = mnew;
      corr[r] = c;
      p0[r] = e0; p1[r] = e1;
    }
#pragma unroll
    for (int dt = 0; dt < 4; ++dt) {
      union { v8f v; float e[8]; } u;
      u.v = oacc[dt];
#pragma unroll
      for (int r = 0; r < 8; ++r) u.e[r] *= corr[r];
      oacc[dt] = u.v;
    }

    // ---- C-layout -> A-layout for P via per-wave LDS (in-order per wave)
#pragma unroll
    for (int r = 0; r < 8; ++r) {
      const int row = r + half * 8;
      Plds[w][row][l16]      = f2bf(p0[r]);
      Plds[w][row][l16 + 16] = f2bf(p1[r]);
    }
    __builtin_amdgcn_wave_barrier();
    const __bf16* pr = &Plds[w][l16][half * 16];
    v16bf pa = cat8(*(const v8bf*)pr, *(const v8bf*)(pr + 8));
    __builtin_amdgcn_wave_barrier();

    // ---- O += P * V  (V pre-transposed: [b][h][d][t], contiguous along t)
#pragma unroll
    for (int dt = 0; dt < 4; ++dt) {
      const __bf16* pv = Vt +
          (size_t)((b * HEADS_ + h) * HD_ + dt * 16 + l16) * SEQ_ +
          t0 + half * 16;
      v16bf vf = *(const v16bf*)pv;
      oacc[dt] = wmma_bf16(pa, vf, oacc[dt]);
    }
  }

  // ---- epilogue: O /= l, store bf16
  float inv[8];
#pragma unroll
  for (int r = 0; r < 8; ++r) inv[r] = 1.0f / lrow[r];
#pragma unroll
  for (int dt = 0; dt < 4; ++dt) {
    union { v8f v; float e[8]; } u;
    u.v = oacc[dt];
#pragma unroll
    for (int r = 0; r < 8; ++r) {
      const int row = m0 + r + half * 8;
      Y[(size_t)(b * SEQ_ + row) * CDIM_ + h * HD_ + dt * 16 + l16] =
          f2bf(u.e[r] * inv[r]);
    }
  }
}

// ---------------------------------------------------------------- launch
extern "C" void kernel_launch(void* const* d_in, const int* in_sizes, int n_in,
                              void* d_out, int out_size, void* d_ws,
                              size_t ws_size, hipStream_t stream) {
  const float* x   = (const float*)d_in[0];
  const float* enc = (const float*)d_in[1];
  const float* Wq  = (const float*)d_in[2];
  const float* bq  = (const float*)d_in[3];
  const float* Wk  = (const float*)d_in[4];
  const float* bk  = (const float*)d_in[5];
  const float* Wv  = (const float*)d_in[6];
  const float* bv  = (const float*)d_in[7];
  const float* Wp  = (const float*)d_in[8];
  const float* bp  = (const float*)d_in[9];
  float* out = (float*)d_out;

  const unsigned nTok = BATCH_ * SEQ_ * CDIM_;  // 8388608
  const unsigned nW   = CDIM_ * CDIM_;          // 1048576

  __bf16* ws  = (__bf16*)d_ws;
  __bf16* xb  = ws;                       // 8388608
  __bf16* eb  = ws + 8388608u;            // 8388608
  __bf16* wqb = ws + 16777216u;           // 4 x 1048576
  __bf16* wkb = wqb + nW;
  __bf16* wvb = wkb + nW;
  __bf16* wpb = wvb + nW;
  __bf16* qb  = ws + 20971520u;           // 8388608
  __bf16* kb  = ws + 29360128u;           // 8388608
  __bf16* vb  = ws + 37748736u;           // 8388608
  __bf16* vt  = xb;                       // reuse: x dead after Q gemm
  __bf16* yb  = eb;                       // reuse: enc dead after K/V gemms

  const dim3 blk(256);
  cvt_f32_to_bf16_kernel<<<(nTok + 255) / 256, blk, 0, stream>>>(x, xb, nTok);
  cvt_f32_to_bf16_kernel<<<(nTok + 255) / 256, blk, 0, stream>>>(enc, eb, nTok);
  cvt_f32_to_bf16_kernel<<<(nW + 255) / 256, blk, 0, stream>>>(Wq, wqb, nW);
  cvt_f32_to_bf16_kernel<<<(nW + 255) / 256, blk, 0, stream>>>(Wk, wkb, nW);
  cvt_f32_to_bf16_kernel<<<(nW + 255) / 256, blk, 0, stream>>>(Wv, wvb, nW);
  cvt_f32_to_bf16_kernel<<<(nW + 255) / 256, blk, 0, stream>>>(Wp, wpb, nW);

  const dim3 ggrid(BATCH_ * SEQ_ / 128, CDIM_ / 128);  // (64, 8)
  gemm_bf16out_kernel<<<ggrid, blk, 0, stream>>>(xb, wqb, bq, qb, CDIM_, CDIM_);
  gemm_bf16out_kernel<<<ggrid, blk, 0, stream>>>(eb, wkb, bk, kb, CDIM_, CDIM_);
  gemm_bf16out_kernel<<<ggrid, blk, 0, stream>>>(eb, wvb, bv, vb, CDIM_, CDIM_);

  transpose_v_kernel<<<(nTok + 255) / 256, blk, 0, stream>>>(vb, vt);

  // one wave per (b, h, m-tile): 4*16*128 = 8192 waves -> 1024 blocks
  attn_kernel<<<BATCH_ * HEADS_ * (SEQ_ / 16) / 8, blk, 0, stream>>>(qb, kb, vt,
                                                                     yb);

  gemm_f32out_kernel<<<ggrid, blk, 0, stream>>>(yb, wpb, bp, out, CDIM_, CDIM_);
}
